// S5Block_35553739276312
// MI455X (gfx1250) — compile-verified, matrix operand order checked by reading, not run on hardware
//
#include <hip/hip_runtime.h>
#include <hip/hip_bf16.h>

// ---------------------------------------------------------------------------
// S5 block for MI455X (gfx1250, wave32, WMMA + async-to-LDS staging).
// Pipeline:
//   K0 prep    : lambda_bar, B_bar, C-combined, w2 -> bf16 swizzled fragments
//   K1 ln      : LayerNorm rows -> xn (bf16)
//   K2 gemm    : Bu[L,256] = xn[L,512] x Bbar[512,256]     (v_wmma bf16->f32)
//   K3 scan    : local scans over 256 chunks of 128 (in-place on Bu, f32)
//   K4 carries : 128-thread serial scan of chunk carries (lambda^128)
//   K5 fixup   : xs = local + lambda^{t+1} * prefix -> bf16
//   K6 gemm    : ys = xs[L,256] x C[256,512]; y = ys + d*xn; x1 = gelu(y)
//   K7 gemm    : z = x1[L,512] x w2[512,512]; out = x + y*sigmoid(z+b2)
// GEMMs: one block = 4 waves sharing one 32-row A panel. The panel is staged
// into LDS via GLOBAL_LOAD_ASYNC_TO_LDS_B128 (ASYNCcnt), then each wave does
// 2x2 register-blocked WMMA: A-fragments from LDS, B-fragments from global.
// ---------------------------------------------------------------------------

#define L_  32768
#define H_  512
#define P_  128
#define N1_ 256      // 2*P (re || im planes)
#define NC_ 256      // scan chunks
#define CL_ 128      // chunk length (L / NC)

typedef __attribute__((ext_vector_type(16))) __bf16 bf16x16;
typedef __attribute__((ext_vector_type(8)))  __bf16 bf16x8;
typedef __attribute__((ext_vector_type(8)))  float  f32x8;

// ---- math helpers ----------------------------------------------------------
__device__ __forceinline__ float sigmoidf_(float v) {
  return 1.0f / (1.0f + __expf(-v));
}
__device__ __forceinline__ float geluf_(float v) {
  const float c = 0.7978845608028654f;
  float t = c * (v + 0.044715f * v * v * v);
  return 0.5f * v * (1.0f + tanhf(t));
}

// ---- CDNA5 async global->LDS copy (ASYNCcnt) -------------------------------
__device__ __forceinline__ void async_copy_b128(unsigned int lds_off,
                                                const __bf16* g) {
  asm volatile("global_load_async_to_lds_b128 %0, %1, off"
               :: "v"(lds_off), "v"(g)
               : "memory");
}
__device__ __forceinline__ void wait_asynccnt0() {
  asm volatile("s_wait_asynccnt 0x0" ::: "memory");
}

// ---- WMMA B-fragment swizzle ----------------------------------------------
// Logical [K][N] row-major element (k,n) -> fragment-major storage so each
// lane's 16 B-elements of a 32x16 tile are one contiguous 32B run.
__device__ __forceinline__ size_t swz(int k, int n, int N) {
  int kt = k >> 5, kin = k & 31;
  int nt = n >> 4, nin = n & 15;
  int half = (kin >> 3) & 1;
  int e = (kin & 7) + ((kin & 16) ? 8 : 0);
  int lane = half * 16 + nin;
  return ((((size_t)kt * (size_t)(N >> 4) + (size_t)nt) * 32) + (size_t)lane) * 16 + (size_t)e;
}

// ---- fragment loads --------------------------------------------------------
__device__ __forceinline__ bf16x16 load_afrag(const __bf16* arow, int k0, int half) {
  bf16x8 a0 = *(const bf16x8*)(arow + k0 + 8 * half);        // K: k0 + {0..7}+8h
  bf16x8 a1 = *(const bf16x8*)(arow + k0 + 16 + 8 * half);   // K: k0+16+{0..7}+8h
  return __builtin_shufflevector(a0, a1, 0, 1, 2, 3, 4, 5, 6, 7,
                                         8, 9, 10, 11, 12, 13, 14, 15);
}
__device__ __forceinline__ bf16x16 load_bfrag(const __bf16* __restrict__ Bsw,
                                              int kt, int tileN, int ntN, int lane) {
  return *(const bf16x16*)(Bsw +
      (((size_t)kt * (size_t)ntN + (size_t)tileN) * 32 + (size_t)lane) * 16);
}

// ---- GEMM core: async-stage 32xKK A panel to LDS, 2x2 WMMA per wave --------
template<int KK>
__device__ __forceinline__ void gemm_core_lds(const __bf16* __restrict__ A,
                                              const __bf16* __restrict__ Bsw,
                                              __bf16* ldsA,
                                              int tm2, int tn2, int ntN,
                                              f32x8 acc[4]) {
  // A panel is 32 contiguous rows of KK bf16 (lda == KK for all our GEMMs).
  const __bf16* Apanel = A + (size_t)tm2 * 32 * KK;
  constexpr int CHUNKS = (32 * KK) / 8;   // 16B chunks; multiple of 128
#pragma unroll 4
  for (int c = threadIdx.x; c < CHUNKS; c += 128) {
    unsigned int loff = (unsigned int)(uintptr_t)(ldsA + c * 8);
    async_copy_b128(loff, Apanel + c * 8);
  }
  wait_asynccnt0();
  __syncthreads();

  int lane = threadIdx.x & 31;
  int half = lane >> 4;
  const __bf16* arow0 = ldsA + (size_t)(lane & 15) * KK;
  const __bf16* arow1 = arow0 + (size_t)16 * KK;
  int tn0 = tn2 * 2, tn1 = tn0 + 1;
#pragma unroll
  for (int k0 = 0; k0 < KK; k0 += 32) {
    bf16x16 av0 = load_afrag(arow0, k0, half);   // ds_load from LDS
    bf16x16 av1 = load_afrag(arow1, k0, half);
    bf16x16 bv0 = load_bfrag(Bsw, k0 >> 5, tn0, ntN, lane);
    bf16x16 bv1 = load_bfrag(Bsw, k0 >> 5, tn1, ntN, lane);
    acc[0] = __builtin_amdgcn_wmma_f32_16x16x32_bf16(false, av0, false, bv0,
                                                     (short)0, acc[0], false, false);
    acc[1] = __builtin_amdgcn_wmma_f32_16x16x32_bf16(false, av0, false, bv1,
                                                     (short)0, acc[1], false, false);
    acc[2] = __builtin_amdgcn_wmma_f32_16x16x32_bf16(false, av1, false, bv0,
                                                     (short)0, acc[2], false, false);
    acc[3] = __builtin_amdgcn_wmma_f32_16x16x32_bf16(false, av1, false, bv1,
                                                     (short)0, acc[3], false, false);
  }
}

// ---- K0: parameter preparation --------------------------------------------
__global__ void prep_params(const float* __restrict__ lr, const float* __restrict__ li,
                            const float* __restrict__ bmat, const float* __restrict__ cmat,
                            const float* __restrict__ delta, const float* __restrict__ w2,
                            float* __restrict__ lamws, __bf16* __restrict__ BbarSw,
                            __bf16* __restrict__ CmatSw, __bf16* __restrict__ w2Sw) {
  int tid = blockIdx.x * blockDim.x + threadIdx.x;  // covers H*H = 262144

  if (tid < H_ * H_) {
    int k = tid / H_, n = tid % H_;
    w2Sw[swz(k, n, H_)] = (__bf16)w2[tid];
  }
  if (tid < P_) {
    float s  = __expf(delta[tid]);
    float hr = 0.5f * s * lr[tid], hi = 0.5f * s * li[tid];
    float dr = 1.0f - hr;
    float den = dr * dr + hi * hi;
    float blr = dr / den, bli = hi / den;  // bl = 1/(dr - i*hi) = (dr + i*hi)/den
    float nr = 1.0f + hr, ni = hi;
    lamws[tid]       = blr * nr - bli * ni;
    lamws[P_ + tid]  = blr * ni + bli * nr;
  }
  if (tid < H_ * N1_) {  // B_bar, logical [k=h][n], n<128 -> re plane, else im
    int h = tid / N1_, n = tid % N1_;
    int p = n & (P_ - 1);
    float s  = __expf(delta[p]);
    float hr = 0.5f * s * lr[p], hi = 0.5f * s * li[p];
    float dr = 1.0f - hr;
    float den = dr * dr + hi * hi;
    float scr = s * dr / den, sci = s * hi / den;  // bl*step
    float bre = bmat[((size_t)p * H_ + h) * 2 + 0];
    float bim = bmat[((size_t)p * H_ + h) * 2 + 1];
    float vre = scr * bre - sci * bim;
    float vim = scr * bim + sci * bre;
    BbarSw[swz(h, n, N1_)] = (__bf16)((n < P_) ? vre : vim);
  }
  if (tid < N1_ * H_) {  // C combined: row k<128 -> 2*C_re, k>=128 -> -2*C_im
    int k = tid / H_, h = tid % H_;
    int p = k & (P_ - 1);
    float v = (k < P_) ?  2.0f * cmat[((size_t)h * P_ + p) * 2 + 0]
                       : -2.0f * cmat[((size_t)h * P_ + p) * 2 + 1];
    CmatSw[swz(k, h, H_)] = (__bf16)v;
  }
}

// ---- K1: LayerNorm ---------------------------------------------------------
__global__ void ln_kernel(const float* __restrict__ x, const float* __restrict__ g,
                          const float* __restrict__ beta, __bf16* __restrict__ xn) {
  int row = blockIdx.x;
  const float* xr = x + (size_t)row * H_;
  int t = threadIdx.x;  // 256 threads, 2 elems each
  float v0 = xr[t], v1 = xr[t + 256];
  float s = v0 + v1, sq = v0 * v0 + v1 * v1;
  for (int off = 16; off; off >>= 1) {
    s  += __shfl_down(s,  off, 32);
    sq += __shfl_down(sq, off, 32);
  }
  __shared__ float sh[16];
  __shared__ float mrs[2];
  int wave = t >> 5, lane = t & 31;
  if (lane == 0) { sh[wave] = s; sh[8 + wave] = sq; }
  __syncthreads();
  if (t == 0) {
    float S = 0.0f, Q = 0.0f;
    for (int i = 0; i < 8; ++i) { S += sh[i]; Q += sh[8 + i]; }
    float mean = S * (1.0f / H_);
    float var  = Q * (1.0f / H_) - mean * mean;
    mrs[0] = mean;
    mrs[1] = rsqrtf(var + 1e-5f);
  }
  __syncthreads();
  float mean = mrs[0], rstd = mrs[1];
  xn[(size_t)row * H_ + t]       = (__bf16)(((v0 - mean) * rstd) * g[t]       + beta[t]);
  xn[(size_t)row * H_ + t + 256] = (__bf16)(((v1 - mean) * rstd) * g[t + 256] + beta[t + 256]);
}

// ---- K2: Bu = xn x B_bar ---------------------------------------------------
// grid.x = (L/32) * 2 ; block owns tm2 and 4 consecutive tn2 (ntN2 = 8)
__global__ void gemm_bu(const __bf16* __restrict__ xn, const __bf16* __restrict__ Bsw,
                        float* __restrict__ Bu) {
  __shared__ __bf16 ldsA[32 * H_];
  int tm2 = blockIdx.x >> 1;
  int tn2 = (blockIdx.x & 1) * 4 + (threadIdx.x >> 5);
  f32x8 acc[4] = {{}, {}, {}, {}};
  gemm_core_lds<H_>(xn, Bsw, ldsA, tm2, tn2, N1_ / 16, acc);
  int lane  = threadIdx.x & 31;
  int col0  = tn2 * 32 + (lane & 15);
  int row0  = tm2 * 32 + (lane >> 4) * 8;
#pragma unroll
  for (int i = 0; i < 2; ++i) {
#pragma unroll
    for (int j = 0; j < 2; ++j) {
      f32x8 a = acc[i * 2 + j];
#pragma unroll
      for (int r = 0; r < 8; ++r)
        Bu[(size_t)(row0 + i * 16 + r) * N1_ + (col0 + j * 16)] = a[r];
    }
  }
}

// ---- K3: local chunk scans (in place on Bu) --------------------------------
__global__ void scan_local(const float* __restrict__ lamws, float* __restrict__ Bu,
                           float* __restrict__ carries) {
  int p = threadIdx.x;        // 0..127
  int chunk = blockIdx.x;     // 0..NC-1
  float ar = lamws[p], ai = lamws[P_ + p];
  float sr = 0.0f, si = 0.0f;
  float* base = Bu + (size_t)chunk * CL_ * N1_;
  for (int t = 0; t < CL_; ++t) {
    float br = base[t * N1_ + p], bi = base[t * N1_ + P_ + p];
    float nr = ar * sr - ai * si + br;
    float ni = ar * si + ai * sr + bi;
    sr = nr; si = ni;
    base[t * N1_ + p] = sr;
    base[t * N1_ + P_ + p] = si;
  }
  carries[chunk * N1_ + p]       = sr;
  carries[chunk * N1_ + P_ + p]  = si;
}

// ---- K4: scan the chunk carries (1 block, 128 threads) ---------------------
__global__ void scan_carries(const float* __restrict__ lamws,
                             const float* __restrict__ carries,
                             float* __restrict__ prefix) {
  int p = threadIdx.x;
  float ar = lamws[p], ai = lamws[P_ + p];
  float cr = 1.0f, ci = 0.0f;                 // lambda^CL
  for (int i = 0; i < CL_; ++i) {
    float nr = cr * ar - ci * ai;
    ci = cr * ai + ci * ar;
    cr = nr;
  }
  float sr = 0.0f, si = 0.0f;
  for (int c = 0; c < NC_; ++c) {
    prefix[c * N1_ + p]      = sr;
    prefix[c * N1_ + P_ + p] = si;
    float br = carries[c * N1_ + p], bi = carries[c * N1_ + P_ + p];
    float nr = cr * sr - ci * si + br;
    float ni = cr * si + ci * sr + bi;
    sr = nr; si = ni;
  }
}

// ---- K5: fix-up + convert to bf16 ------------------------------------------
__global__ void scan_fixup(const float* __restrict__ lamws, const float* __restrict__ Bu,
                           const float* __restrict__ prefix, __bf16* __restrict__ xs) {
  int p = threadIdx.x;
  int chunk = blockIdx.x;
  float ar = lamws[p], ai = lamws[P_ + p];
  float cr = prefix[chunk * N1_ + p], ci = prefix[chunk * N1_ + P_ + p];
  const float* base = Bu + (size_t)chunk * CL_ * N1_;
  __bf16* xbase = xs + (size_t)chunk * CL_ * N1_;
  for (int t = 0; t < CL_; ++t) {
    float nr = ar * cr - ai * ci;   // corr = lambda^{t+1} * prefix_state
    float ni = ar * ci + ai * cr;
    cr = nr; ci = ni;
    xbase[t * N1_ + p]      = (__bf16)(base[t * N1_ + p]      + cr);
    xbase[t * N1_ + P_ + p] = (__bf16)(base[t * N1_ + P_ + p] + ci);
  }
}

// ---- K6: ys = xs x C ; y = ys + d*xn ; x1 = gelu(y) ------------------------
// grid.x = (L/32) * 4 ; block owns tm2 and 4 consecutive tn2 (ntN2 = 16)
__global__ void gemm_y(const __bf16* __restrict__ xs, const __bf16* __restrict__ Csw,
                       const __bf16* __restrict__ xnbf, const float* __restrict__ dvec,
                       float* __restrict__ y, __bf16* __restrict__ x1) {
  __shared__ __bf16 ldsA[32 * N1_];
  int tm2 = blockIdx.x >> 2;
  int tn2 = (blockIdx.x & 3) * 4 + (threadIdx.x >> 5);
  f32x8 acc[4] = {{}, {}, {}, {}};
  gemm_core_lds<N1_>(xs, Csw, ldsA, tm2, tn2, H_ / 16, acc);
  int lane = threadIdx.x & 31;
  int col0 = tn2 * 32 + (lane & 15);
  int row0 = tm2 * 32 + (lane >> 4) * 8;
#pragma unroll
  for (int i = 0; i < 2; ++i) {
#pragma unroll
    for (int j = 0; j < 2; ++j) {
      f32x8 a = acc[i * 2 + j];
      int col = col0 + j * 16;
      float dh = dvec[col];
#pragma unroll
      for (int r = 0; r < 8; ++r) {
        size_t idx = (size_t)(row0 + i * 16 + r) * H_ + col;
        float yv = a[r] + dh * (float)xnbf[idx];
        y[idx]  = yv;
        x1[idx] = (__bf16)geluf_(yv);
      }
    }
  }
}

// ---- K7: z = x1 x w2 ; out = x + y*sigmoid(z + b2) -------------------------
__global__ void gemm_out(const __bf16* __restrict__ x1, const __bf16* __restrict__ w2sw,
                         const float* __restrict__ y, const float* __restrict__ x,
                         const float* __restrict__ b2, float* __restrict__ out) {
  __shared__ __bf16 ldsA[32 * H_];
  int tm2 = blockIdx.x >> 2;
  int tn2 = (blockIdx.x & 3) * 4 + (threadIdx.x >> 5);
  f32x8 acc[4] = {{}, {}, {}, {}};
  gemm_core_lds<H_>(x1, w2sw, ldsA, tm2, tn2, H_ / 16, acc);
  int lane = threadIdx.x & 31;
  int col0 = tn2 * 32 + (lane & 15);
  int row0 = tm2 * 32 + (lane >> 4) * 8;
#pragma unroll
  for (int i = 0; i < 2; ++i) {
#pragma unroll
    for (int j = 0; j < 2; ++j) {
      f32x8 a = acc[i * 2 + j];
      int col = col0 + j * 16;
      float bb = b2[col];
#pragma unroll
      for (int r = 0; r < 8; ++r) {
        size_t idx = (size_t)(row0 + i * 16 + r) * H_ + col;
        out[idx] = x[idx] + y[idx] * sigmoidf_(a[r] + bb);
      }
    }
  }
}

// ---------------------------------------------------------------------------
extern "C" void kernel_launch(void* const* d_in, const int* in_sizes, int n_in,
                              void* d_out, int out_size, void* d_ws, size_t ws_size,
                              hipStream_t stream) {
  const float* x   = (const float*)d_in[0];
  const float* lr  = (const float*)d_in[1];
  const float* li  = (const float*)d_in[2];
  const float* bm  = (const float*)d_in[3];
  const float* cm  = (const float*)d_in[4];
  const float* dv  = (const float*)d_in[5];
  const float* dl  = (const float*)d_in[6];
  const float* g   = (const float*)d_in[7];
  const float* be  = (const float*)d_in[8];
  const float* w2  = (const float*)d_in[9];
  const float* b2  = (const float*)d_in[10];
  float* out = (float*)d_out;

  char* ws = (char*)d_ws;
  size_t o = 0;
  auto take = [&](size_t bytes) -> char* {
    char* r = ws + o;
    o += (bytes + 255) & ~(size_t)255;
    return r;
  };
  float*  lamws   = (float*) take(2 * P_ * sizeof(float));
  __bf16* BbarSw  = (__bf16*)take((size_t)H_ * N1_ * 2);
  __bf16* CmatSw  = (__bf16*)take((size_t)N1_ * H_ * 2);
  __bf16* w2Sw    = (__bf16*)take((size_t)H_ * H_ * 2);
  __bf16* xnbf    = (__bf16*)take((size_t)L_ * H_ * 2);
  float*  Bu      = (float*) take((size_t)L_ * N1_ * 4);
  float*  carries = (float*) take((size_t)NC_ * N1_ * 4);
  float*  prefix  = (float*) take((size_t)NC_ * N1_ * 4);
  __bf16* xsbf    = (__bf16*)take((size_t)L_ * N1_ * 2);
  float*  ybuf    = (float*) take((size_t)L_ * H_ * 4);
  __bf16* x1bf    = (__bf16*)take((size_t)L_ * H_ * 2);

  // K0: parameter prep (262144 threads)
  prep_params<<<(H_ * H_) / 256, 256, 0, stream>>>(lr, li, bm, cm, dl, w2,
                                                   lamws, BbarSw, CmatSw, w2Sw);
  // K1: LayerNorm
  ln_kernel<<<L_, 256, 0, stream>>>(x, g, be, xnbf);
  // K2: Bu GEMM — block = 4 waves sharing one 32-row A panel
  gemm_bu<<<(L_ / 32) * 2, 128, 0, stream>>>(xnbf, BbarSw, Bu);
  // K3-K5: chunked associative scan
  scan_local<<<NC_, P_, 0, stream>>>(lamws, Bu, carries);
  scan_carries<<<1, P_, 0, stream>>>(lamws, carries, prefix);
  scan_fixup<<<NC_, P_, 0, stream>>>(lamws, Bu, prefix, xsbf);
  // K6: ys GEMM + D-term + gelu
  gemm_y<<<(L_ / 32) * 4, 128, 0, stream>>>(xsbf, CmatSw, xnbf, dv, ybuf, x1bf);
  // K7: glu GEMM + residual
  gemm_out<<<(L_ / 32) * 4, 128, 0, stream>>>(x1bf, w2Sw, ybuf, x, b2, out);
}
